// FEBdetector_21174188769470
// MI455X (gfx1250) — matrix operand order repaired and checked by reading, not compiled, not run on hardware
//
#include <hip/hip_runtime.h>
#include <hip/hip_bf16.h>
#include <cstdint>

typedef __attribute__((ext_vector_type(16))) __bf16 v16bf;
typedef __attribute__((ext_vector_type(8)))  __bf16 v8bf;
typedef __attribute__((ext_vector_type(8)))  float  v8f;

#define B_ 256
#define L_ 105
#define M_ 55
#define D_ 256

// ---- WMMA fragment loaders (bf16, 16x16x32, row-major sources) ----
// A-matrix 16x32 layout: lanes0-15 row m, K = 0..15; lanes16-31 row m, K = 16..31
__device__ __forceinline__ v16bf load_a_frag(const __bf16* base, int ld, int row, int k0, int lane) {
  const __bf16* p = base + (size_t)row * ld + k0 + ((lane >> 4) << 4);
  v8bf lo = *(const v8bf*)p;
  v8bf hi = *(const v8bf*)(p + 8);
  return __builtin_shufflevector(lo, hi, 0,1,2,3,4,5,6,7,8,9,10,11,12,13,14,15);
}
// B-matrix 32x16: lanes0-15 hold K=0..15 of col n, lanes16-31 K=16..31.  Source is Bt (N x K row-major).
__device__ __forceinline__ v16bf load_b_frag(const __bf16* base, int ld, int row, int k0, int lane) {
  const __bf16* p = base + (size_t)row * ld + k0 + ((lane >> 4) << 4);
  return *(const v16bf*)p;
}

// ---------------- RevIN ----------------
__global__ void revin_kernel(const float* __restrict__ x, const float* __restrict__ w_rev,
                             const float* __restrict__ b_rev, float* __restrict__ xn) {
  int b = blockIdx.x, m = threadIdx.x;
  if (m >= M_) return;
  const float* xb = x + (size_t)b * L_ * M_;
  float s = 0.f, s2 = 0.f;
  for (int l = 0; l < L_; ++l) { float v = xb[l * M_ + m]; s += v; s2 += v * v; }
  float mu = s * (1.0f / L_);
  float var = s2 * (1.0f / L_) - mu * mu;
  float inv = rsqrtf(fmaxf(var, 0.f) + 1e-5f);
  float w = w_rev[m], bb = b_rev[m];
  float* ob = xn + (size_t)b * L_ * M_;
  for (int l = 0; l < L_; ++l) ob[l * M_ + m] = (xb[l * M_ + m] - mu) * inv * w + bb;
}

// ---------------- Gather circular-conv activations into bf16 GEMM-A ----------------
template<int P>
__global__ void pack_patch_kernel(const float* __restrict__ xn, __bf16* __restrict__ Ap) {
  constexpr int n = L_ / P;
  constexpr int Kreal = 3 * M_ * P;
  constexpr int Kpad = ((Kreal + 31) / 32) * 32;
  size_t idx = (size_t)blockIdx.x * 256 + threadIdx.x;
  size_t total = (size_t)B_ * n * Kpad;
  if (idx >= total) return;
  int col = (int)(idx % Kpad);
  int row = (int)(idx / Kpad);
  float v = 0.0f;
  if (col < Kreal) {
    int kk = col / (M_ * P);
    int rem = col - kk * (M_ * P);
    int m = rem / P;
    int j = rem - m * P;
    int tn = row % n;
    int b  = row / n;
    int ts = ((tn + kk - 1 + n) % n) * P + j;        // circular pad by 1 token
    v = xn[((size_t)b * L_ + ts) * M_ + m];
  }
  Ap[idx] = (__bf16)v;
}

// ---------------- conv weight (D,Cin,3) -> bf16 Wt[D][kk*Cin+c] padded ----------------
__global__ void pack_convw_kernel(const float* __restrict__ w, __bf16* __restrict__ Wt,
                                  int Cin, int Kpad) {
  int idx = blockIdx.x * 256 + threadIdx.x;
  if (idx >= 256 * Kpad) return;
  int d = idx / Kpad;
  int col = idx - d * Kpad;
  float v = 0.0f;
  if (col < 3 * Cin) {
    int kk = col / Cin;
    int c = col - kk * Cin;
    v = w[(d * Cin + c) * 3 + kk];
  }
  Wt[idx] = (__bf16)v;
}

// ---------------- (256,256) f32 -> bf16 transpose ----------------
__global__ void pack_wT_kernel(const float* __restrict__ W, __bf16* __restrict__ Wt) {
  int idx = blockIdx.x * 256 + threadIdx.x;   // 65536
  int nn = idx >> 8, kk = idx & 255;
  Wt[idx] = (__bf16)W[(kk << 8) + nn];
}

// ---------------- generic C = A(Mr x K) * Bt^T(K x 256), bf16 in/out, f32 acc ----------------
// Each wave computes a 16x64 strip: A fragment loaded once per k-step, reused for 4 WMMAs.
// epi: 0 = plain, 1 = +sinusoidal pos-embed (tokens of length T), 2 = +bias[col]
__global__ void gemm_bt_kernel(const __bf16* __restrict__ A, const __bf16* __restrict__ Bt,
                               __bf16* __restrict__ C, const float* __restrict__ bias,
                               int Mr, int K, int T, int epi) {
  int lane = threadIdx.x & 31;
  int tile = blockIdx.x * 4 + (threadIdx.x >> 5);
  int tiles = (Mr >> 4) << 2;                 // rowTiles * 4 col-groups (of 64)
  if (tile >= tiles) return;
  int tr = tile >> 2, cg = tile & 3;
  int row0 = tr << 4, col0 = cg << 6;
  const __bf16* aP = A + (size_t)(row0 + (lane & 15)) * K + ((lane >> 4) << 4);
  const __bf16* bP = Bt + (size_t)(col0 + (lane & 15)) * K + ((lane >> 4) << 4);
  size_t bStep = (size_t)16 * K;
  v8f accs[4] = {};
  for (int k0 = 0; k0 < K; k0 += 32) {
    v8bf lo = *(const v8bf*)(aP + k0);
    v8bf hi = *(const v8bf*)(aP + k0 + 8);
    v16bf a = __builtin_shufflevector(lo, hi, 0,1,2,3,4,5,6,7,8,9,10,11,12,13,14,15);
    v16bf b0 = *(const v16bf*)(bP + k0);
    v16bf b1 = *(const v16bf*)(bP + bStep + k0);
    v16bf b2 = *(const v16bf*)(bP + 2 * bStep + k0);
    v16bf b3 = *(const v16bf*)(bP + 3 * bStep + k0);
    accs[0] = __builtin_amdgcn_wmma_f32_16x16x32_bf16(false, a, false, b0, (short)0, accs[0], false, false);
    accs[1] = __builtin_amdgcn_wmma_f32_16x16x32_bf16(false, a, false, b1, (short)0, accs[1], false, false);
    accs[2] = __builtin_amdgcn_wmma_f32_16x16x32_bf16(false, a, false, b2, (short)0, accs[2], false, false);
    accs[3] = __builtin_amdgcn_wmma_f32_16x16x32_bf16(false, a, false, b3, (short)0, accs[3], false, false);
  }
  int rbase = row0 + ((lane >> 4) << 3);
  int tbase = (epi == 1) ? (rbase % T) : 0;   // token index of first row, wraps within tile
#pragma unroll
  for (int j = 0; j < 4; ++j) {
    int col = col0 + (j << 4) + (lane & 15);
    float pe_div = (epi == 1) ? __expf((float)(col & ~1) * (-0.03597789244f)) : 0.0f; // -ln(1e4)/256
#pragma unroll
    for (int r = 0; r < 8; ++r) {
      int row = rbase + r;
      float v = accs[j][r];
      if (epi == 1) {
        int t = tbase + r; if (t >= T) t -= T;
        float ang = (float)t * pe_div;
        v += (col & 1) ? __cosf(ang) : __sinf(ang);
      } else if (epi == 2) {
        v += bias[col];
      }
      C[(size_t)row * D_ + col] = (__bf16)v;
    }
  }
}

// ---------------- Mahalanobis mask: per-batch T x T, WMMA Gram + fused normalize ----------------
__global__ __launch_bounds__(256) void maha_mask_kernel(const __bf16* __restrict__ XA,
                                                        const float* __restrict__ sq,
                                                        float* __restrict__ mask, int T, int TP) {
  __shared__ float s_tile[112 * 112];
  __shared__ float s_max[112];
  int b = blockIdx.x;
  int tid = threadIdx.x, lane = tid & 31, wid = tid >> 5;
  const __bf16* Xb = XA + (size_t)b * T * D_;
  const float* sqb = sq + (size_t)b * T;
  int nT = TP >> 4;
  for (int t = wid; t < nT * nT; t += 8) {
    int tr = t / nT, tc = t - tr * nT;
    int ar = tr * 16 + (lane & 15); if (ar >= T) ar = T - 1;
    int br = tc * 16 + (lane & 15); if (br >= T) br = T - 1;
    v8f acc = {};
    for (int k0 = 0; k0 < D_; k0 += 32) {
      v16bf a  = load_a_frag(Xb, D_, ar, k0, lane);
      v16bf bb = load_b_frag(Xb, D_, br, k0, lane);
      acc = __builtin_amdgcn_wmma_f32_16x16x32_bf16(false, a, false, bb, (short)0, acc, false, false);
    }
    int rbase = tr * 16 + ((lane >> 4) << 3);
    int col = tc * 16 + (lane & 15);
#pragma unroll
    for (int r = 0; r < 8; ++r) {
      int row = rbase + r;
      int rc = row < T ? row : T - 1;
      int cc = col < T ? col : T - 1;
      float d2 = sqb[rc] + sqb[cc] - 2.0f * acc[r];
      float inv = (row == col || row >= T || col >= T) ? 0.0f
                                                       : 1.0f / (fmaxf(d2, 0.0f) + 1e-10f);
      s_tile[row * TP + col] = inv;
    }
  }
  __syncthreads();
  for (int i = tid; i < T; i += 256) {
    const float* rowp = &s_tile[i * TP];
    float m = 0.0f;
    for (int j = 0; j < T; ++j) m = fmaxf(m, rowp[j]);
    s_max[i] = fmaxf(m, 1e-30f);
  }
  __syncthreads();
  float* mb = mask + (size_t)b * T * T;
  for (int i = 0; i < T; ++i) {
    float rm = 1.0f / s_max[i];
    const float* rowp = &s_tile[i * TP];
    for (int j = tid; j < T; j += 256)
      mb[i * T + j] = (i == j) ? 0.99f : 0.99f * rowp[j] * rm;
  }
}

// ---------------- attention map: scores via WMMA, softmax*mask, upsample-by-P write ----------------
template<int P>
__global__ __launch_bounds__(256) void attn_kernel(const __bf16* __restrict__ q,
                                                   const __bf16* __restrict__ k,
                                                   const float* __restrict__ mask,
                                                   float* __restrict__ out, int T, int TP) {
  __shared__ float s_tile[112 * 112];
  int b = blockIdx.x;
  int tid = threadIdx.x, lane = tid & 31, wid = tid >> 5;
  const __bf16* qb = q + (size_t)b * T * D_;
  const __bf16* kb = k + (size_t)b * T * D_;
  int nT = TP >> 4;
  for (int t = wid; t < nT * nT; t += 8) {
    int tr = t / nT, tc = t - tr * nT;
    int ar = tr * 16 + (lane & 15); if (ar >= T) ar = T - 1;
    int br = tc * 16 + (lane & 15); if (br >= T) br = T - 1;
    v8f acc = {};
    for (int k0 = 0; k0 < D_; k0 += 32) {
      v16bf a  = load_a_frag(qb, D_, ar, k0, lane);
      v16bf bb = load_b_frag(kb, D_, br, k0, lane);
      acc = __builtin_amdgcn_wmma_f32_16x16x32_bf16(false, a, false, bb, (short)0, acc, false, false);
    }
    int rbase = tr * 16 + ((lane >> 4) << 3);
    int col = tc * 16 + (lane & 15);
#pragma unroll
    for (int r = 0; r < 8; ++r)
      s_tile[(rbase + r) * TP + col] = acc[r] * 0.0625f;   // 1/sqrt(256)
  }
  __syncthreads();
  const float* mbase = mask + (size_t)b * T * T;
  for (int i = tid; i < T; i += 256) {
    float* rowp = &s_tile[i * TP];
    float m = -3.4e38f;
    for (int j = 0; j < T; ++j) m = fmaxf(m, rowp[j]);
    float ssum = 0.0f;
    for (int j = 0; j < T; ++j) ssum += __expf(rowp[j] - m);
    float invs = 1.0f / ssum;
    const float* mr = mbase + i * T;
    for (int j = 0; j < T; ++j) rowp[j] = __expf(rowp[j] - m) * invs * mr[j];
  }
  __syncthreads();
  float* ob = out + (size_t)b * (L_ * L_);
  for (int i = 0; i < L_; ++i) {
    const float* srow = &s_tile[(i / P) * TP];
    for (int j = tid; j < L_; j += 256)
      ob[i * L_ + j] = srow[j / P];
  }
}

// ---------------- row squared norms of bf16 XA ----------------
__global__ void rowsq_kernel(const __bf16* __restrict__ XA, float* __restrict__ sq, int Mr) {
  int r = blockIdx.x * 256 + threadIdx.x;
  if (r >= Mr) return;
  const __bf16* p = XA + (size_t)r * D_;
  float s = 0.0f;
  for (int d = 0; d < D_; ++d) { float v = (float)p[d]; s += v * v; }
  sq[r] = s;
}

extern "C" void kernel_launch(void* const* d_in, const int* in_sizes, int n_in,
                              void* d_out, int out_size, void* d_ws, size_t ws_size,
                              hipStream_t stream) {
  (void)in_sizes; (void)n_in; (void)out_size; (void)ws_size;
  const float* x      = (const float*)d_in[0];
  const float* w_rev  = (const float*)d_in[1];
  const float* b_rev  = (const float*)d_in[2];
  const float* conv_ps[3] = {(const float*)d_in[3], (const float*)d_in[4], (const float*)d_in[5]};
  const float* conv_pn[3] = {(const float*)d_in[6], (const float*)d_in[7], (const float*)d_in[8]};
  const float* A1 = (const float*)d_in[9];
  const float* A2 = (const float*)d_in[10];
  const float* Wq = (const float*)d_in[11];
  const float* bq = (const float*)d_in[12];
  const float* Wk = (const float*)d_in[13];
  const float* bk = (const float*)d_in[14];
  float* outp = (float*)d_out;

  char* ws = (char*)d_ws;
  size_t cur = 0;
  auto alloc = [&](size_t bytes) -> void* {
    void* p = ws + cur;
    cur += (bytes + 255) & ~(size_t)255;
    return p;
  };

  static const int PATCH[3] = {3, 5, 7};
  static const int KPACK[4] = {192, 512, 832, 1184};      // Kpad for p=1,3,5,7
  static const int RPACK[4] = {26880, 8960, 5376, 3840};  // B*n rows

  float*  xnorm = (float*)alloc((size_t)B_ * L_ * M_ * 4);
  __bf16* Apack[4];
  for (int i = 0; i < 4; ++i) Apack[i] = (__bf16*)alloc((size_t)RPACK[i] * KPACK[i] * 2);
  __bf16* WcPN[3]; __bf16* WcPS[3];
  for (int i = 0; i < 3; ++i) WcPN[i] = (__bf16*)alloc((size_t)256 * 192 * 2);
  for (int i = 0; i < 3; ++i) WcPS[i] = (__bf16*)alloc((size_t)256 * KPACK[i + 1] * 2);
  __bf16* WT_A1 = (__bf16*)alloc(65536 * 2);
  __bf16* WT_A2 = (__bf16*)alloc(65536 * 2);
  __bf16* WT_Wq[3]; __bf16* WT_Wk[3];
  for (int l = 0; l < 3; ++l) WT_Wq[l] = (__bf16*)alloc(65536 * 2);
  for (int l = 0; l < 3; ++l) WT_Wk[l] = (__bf16*)alloc(65536 * 2);
  __bf16* xg  = (__bf16*)alloc((size_t)26880 * 256 * 2);
  __bf16* xps = (__bf16*)alloc((size_t)8960 * 256 * 2);
  __bf16* XAb = (__bf16*)alloc((size_t)26880 * 256 * 2);
  float*  sqb = (float*)alloc((size_t)26880 * 4);
  float*  mask_g  = (float*)alloc((size_t)B_ * L_ * L_ * 4);
  float*  mask_ps = (float*)alloc((size_t)B_ * 35 * 35 * 4);
  __bf16* qb = (__bf16*)alloc((size_t)26880 * 256 * 2);
  __bf16* kb = (__bf16*)alloc((size_t)26880 * 256 * 2);

  auto GEMM = [&](const __bf16* A, const __bf16* Bt, __bf16* C, const float* bias,
                  int Mr, int K, int T, int epi) {
    int blocks = Mr >> 4;                     // 4 waves/block, each a 16x64 strip
    gemm_bt_kernel<<<blocks, 128, 0, stream>>>(A, Bt, C, bias, Mr, K, T, epi);
  };

  // -------- prep --------
  revin_kernel<<<B_, 64, 0, stream>>>(x, w_rev, b_rev, xnorm);
  {
    size_t t;
    t = (size_t)RPACK[0] * KPACK[0]; pack_patch_kernel<1><<<(t + 255) / 256, 256, 0, stream>>>(xnorm, Apack[0]);
    t = (size_t)RPACK[1] * KPACK[1]; pack_patch_kernel<3><<<(t + 255) / 256, 256, 0, stream>>>(xnorm, Apack[1]);
    t = (size_t)RPACK[2] * KPACK[2]; pack_patch_kernel<5><<<(t + 255) / 256, 256, 0, stream>>>(xnorm, Apack[2]);
    t = (size_t)RPACK[3] * KPACK[3]; pack_patch_kernel<7><<<(t + 255) / 256, 256, 0, stream>>>(xnorm, Apack[3]);
  }
  for (int i = 0; i < 3; ++i) {
    pack_convw_kernel<<<(256 * 192 + 255) / 256, 256, 0, stream>>>(conv_pn[i], WcPN[i], 55, 192);
    pack_convw_kernel<<<(256 * KPACK[i + 1] + 255) / 256, 256, 0, stream>>>(conv_ps[i], WcPS[i],
                                                                            55 * PATCH[i], KPACK[i + 1]);
  }
  pack_wT_kernel<<<256, 256, 0, stream>>>(A1, WT_A1);
  pack_wT_kernel<<<256, 256, 0, stream>>>(A2, WT_A2);
  for (int l = 0; l < 3; ++l) {
    pack_wT_kernel<<<256, 256, 0, stream>>>(Wq + (size_t)l * 65536, WT_Wq[l]);
    pack_wT_kernel<<<256, 256, 0, stream>>>(Wk + (size_t)l * 65536, WT_Wk[l]);
  }

  const size_t SLICE = (size_t)B_ * L_ * L_;   // per (branch, scale, layer) slab

  // -------- per scale --------
  for (int i = 0; i < 3; ++i) {
    int P = PATCH[i], n = L_ / P, rows = B_ * n, TPs = ((n + 15) / 16) * 16;
    // token embeds (+pos) as gather-GEMMs
    GEMM(Apack[0], WcPN[i], xg, nullptr, 26880, 192, L_, 1);
    GEMM(Apack[i + 1], WcPS[i], xps, nullptr, rows, KPACK[i + 1], n, 1);
    // maha mask (global): XA = x_g @ A2
    GEMM(xg, WT_A2, XAb, nullptr, 26880, 256, 1, 0);
    rowsq_kernel<<<(26880 + 255) / 256, 256, 0, stream>>>(XAb, sqb, 26880);
    maha_mask_kernel<<<B_, 256, 0, stream>>>(XAb, sqb, mask_g, L_, 112);
    // maha mask (patch): XA = x_ps @ A1
    GEMM(xps, WT_A1, XAb, nullptr, rows, 256, 1, 0);
    rowsq_kernel<<<(rows + 255) / 256, 256, 0, stream>>>(XAb, sqb, rows);
    maha_mask_kernel<<<B_, 256, 0, stream>>>(XAb, sqb, mask_ps, n, TPs);

    for (int l = 0; l < 3; ++l) {
      // prior branch (global tokens)
      GEMM(xg, WT_Wq[l], qb, bq + (size_t)l * 256, 26880, 256, 1, 2);
      GEMM(xg, WT_Wk[l], kb, bk + (size_t)l * 256, 26880, 256, 1, 2);
      attn_kernel<1><<<B_, 256, 0, stream>>>(qb, kb, mask_g,
                                             outp + (size_t)(9 + i * 3 + l) * SLICE, L_, 112);
      // series branch (patch tokens, upsampled by P)
      GEMM(xps, WT_Wq[l], qb, bq + (size_t)l * 256, rows, 256, 1, 2);
      GEMM(xps, WT_Wk[l], kb, bk + (size_t)l * 256, rows, 256, 1, 2);
      float* oser = outp + (size_t)(i * 3 + l) * SLICE;
      if (P == 3)      attn_kernel<3><<<B_, 256, 0, stream>>>(qb, kb, mask_ps, oser, n, TPs);
      else if (P == 5) attn_kernel<5><<<B_, 256, 0, stream>>>(qb, kb, mask_ps, oser, n, TPs);
      else             attn_kernel<7><<<B_, 256, 0, stream>>>(qb, kb, mask_ps, oser, n, TPs);
    }
  }
}
